// RelationalGraphConvolution_60954175865082
// MI455X (gfx1250) — matrix operand order repaired and checked by reading, not compiled, not run on hardware
//
#include <hip/hip_runtime.h>
#include <hip/hip_bf16.h>
#include <stddef.h>
#include <stdint.h>

typedef __attribute__((ext_vector_type(16))) __bf16 v16bf;
typedef __attribute__((ext_vector_type(8)))  __bf16 v8bf;
typedef __attribute__((ext_vector_type(4)))  __bf16 v4bf;
typedef __attribute__((ext_vector_type(8)))  float  v8f;
typedef __attribute__((ext_vector_type(4)))  float  v4f;

#define N_ 4096
#define F_ 256
#define U_ 128
#define R_ 16
#define B_ 32
#define BM 64
#define BK 32
#define SPLITK 4
#define RPC (R_ / SPLITK)
#define NTB (N_ / BK)          /* 128 K-tiles per relation */
#define TPC (RPC * NTB)        /* 512 K-tiles per chunk    */

// ---- fragment builders (CDNA5 ISA 7.12.2 layouts, wave32) ----
// A 16x32 bf16: lanes 0-15 -> M=lane, K in {0..7, 16..23}; lanes 16-31 -> M=lane-16, K in {8..15, 24..31}
static __device__ __forceinline__ v16bf afrag_lds(const __bf16* rowbase, int lane) {
  const int kb = (lane & 16) ? 8 : 0;
  v8bf lo = *(const v8bf*)(rowbase + kb);
  v8bf hi = *(const v8bf*)(rowbase + 16 + kb);
  v16bf r;
#pragma unroll
  for (int i = 0; i < 8; ++i) { r[i] = lo[i]; r[i + 8] = hi[i]; }
  return r;
}
// B 32x16 bf16 (stored column-major in LDS, 32 contiguous K per column):
// lanes 0-15 -> N=lane, K in {0..7, 16..23}; lanes 16-31 -> N=lane-16, K in {8..15, 24..31}
static __device__ __forceinline__ v16bf bfrag_lds(const __bf16* colbase, int lane) {
  const int kb = (lane & 16) ? 8 : 0;
  v8bf lo = *(const v8bf*)(colbase + kb);
  v8bf hi = *(const v8bf*)(colbase + 16 + kb);
  v16bf r;
#pragma unroll
  for (int i = 0; i < 8; ++i) { r[i] = lo[i]; r[i + 8] = hi[i]; }
  return r;
}

static __device__ __forceinline__ v4bf cvt4(v4f v) {
  v4bf o;
#pragma unroll
  for (int i = 0; i < 4; ++i) o[i] = (__bf16)v[i];
  return o;
}

// ---- kernel 1: w_rel[r,f,u] = sum_b bases[f,u,b] * coef[r,b] ----
__global__ __launch_bounds__(256) void k_wrel(const float* __restrict__ bases,
                                              const float* __restrict__ coef,
                                              float* __restrict__ wrel) {
  int idx = blockIdx.x * 256 + threadIdx.x;        // idx = r*F*U + f*U + u
  int r  = idx / (F_ * U_);
  int fu = idx - r * (F_ * U_);
  const float* bp = bases + (size_t)fu * B_;
  const float* cp = coef + r * B_;
  float s = 0.f;
#pragma unroll
  for (int b = 0; b < B_; ++b) s += bp[b] * cp[b];
  wrel[idx] = s;
}

// ---- kernel 2: y[r] = x @ W[r] (WMMA); r==R uses self_kernel.
// r<R: write bf16 transposed yT[r][u][m]; r==R: write f32 yself[m][u].
__global__ __launch_bounds__(256) void k_y(const float* __restrict__ x,
                                           const float* __restrict__ wrel,
                                           const float* __restrict__ selfk,
                                           __bf16* __restrict__ yT,
                                           float* __restrict__ yself) {
  __shared__ __bf16 sA[BM * BK];   // 4 KB
  __shared__ __bf16 sB[U_ * BK];   // 8 KB, column-major (K contiguous per u)
  const int m0   = blockIdx.x * BM;
  const int rel  = blockIdx.y;
  const float* W = (rel < R_) ? (wrel + (size_t)rel * F_ * U_) : selfk;
  const int t = threadIdx.x;
  const int wave = t >> 5, lane = t & 31;
  const int msub = wave & 3, ugrp = wave >> 2;
  const int lrow = lane & 15;
  v8f acc[4] = {};

  for (int kk = 0; kk < F_; kk += BK) {
    {  // A tile: x[m0+row][kk+c], f32 -> bf16
      int row = t >> 2, c0 = (t & 3) * 8;
      const float* src = x + (size_t)(m0 + row) * F_ + kk + c0;
      v4f a0 = *(const v4f*)src;
      v4f a1 = *(const v4f*)(src + 4);
      *(v4bf*)&sA[row * BK + c0]     = cvt4(a0);
      *(v4bf*)&sA[row * BK + c0 + 4] = cvt4(a1);
    }
    {  // B tile: W[kk+kl][u] -> sB[u][kl] (transpose into LDS)
      int kl = t >> 3, u0b = (t & 7) * 16;
      const float* src = W + (size_t)(kk + kl) * U_ + u0b;
#pragma unroll
      for (int i = 0; i < 16; ++i) sB[(u0b + i) * BK + kl] = (__bf16)src[i];
    }
    __syncthreads();
    v16bf a = afrag_lds(&sA[(msub * 16 + lrow) * BK], lane);
#pragma unroll
    for (int j = 0; j < 4; ++j) {
      v16bf b = bfrag_lds(&sB[(ugrp * 64 + j * 16 + lrow) * BK], lane);
      acc[j] = __builtin_amdgcn_wmma_f32_16x16x32_bf16(false, a, false, b,
                                                       (short)0, acc[j], false, false);
    }
    __syncthreads();
  }

  if (rel < R_) {
#pragma unroll
    for (int j = 0; j < 4; ++j) {
      int ucol = ugrp * 64 + j * 16 + lrow;
#pragma unroll
      for (int v = 0; v < 8; ++v) {
        int mrow = m0 + msub * 16 + v + ((lane & 16) ? 8 : 0);
        yT[(size_t)rel * U_ * N_ + (size_t)ucol * N_ + mrow] = (__bf16)acc[j][v];
      }
    }
  } else {
#pragma unroll
    for (int j = 0; j < 4; ++j) {
      int ucol = ugrp * 64 + j * 16 + lrow;
#pragma unroll
      for (int v = 0; v < 8; ++v) {
        int mrow = m0 + msub * 16 + v + ((lane & 16) ? 8 : 0);
        yself[(size_t)mrow * U_ + ucol] = acc[j][v];
      }
    }
  }
}

// async bf16 tile copy: global (SGPR64 base + VGPR32 byte offset) -> LDS, 32 bytes/lane
static __device__ __forceinline__ void async_b_copy(const __bf16* base, unsigned goff,
                                                    __bf16* ldsdst) {
  unsigned ldsaddr = (unsigned)(uintptr_t)ldsdst;  // low 32 bits of generic ptr == LDS offset
  unsigned long long sbase = (unsigned long long)(uintptr_t)base;
  asm volatile("global_load_async_to_lds_b128 %0, %1, %2 offset:0\n\t"
               "global_load_async_to_lds_b128 %0, %1, %2 offset:16"
               :: "v"(ldsaddr), "v"(goff), "s"(sbase) : "memory");
}
static __device__ __forceinline__ void wait_async0() {
  asm volatile("s_wait_asynccnt 0x0" ::: "memory");
}

// ---- kernel 3: partial[c] = sum_{r in chunk c} A[r] @ y[r]  (the HBM-bound GEMM) ----
// Double-buffered LDS, software pipelined: A streamed non-temporally through VGPRs
// (needs f32->bf16 cvt), B tile copied global->LDS with GLOBAL_LOAD_ASYNC_TO_LDS_B128.
__global__ __launch_bounds__(256) void k_main(const float* __restrict__ A,
                                              const __bf16* __restrict__ yT,
                                              float* __restrict__ part) {
  __shared__ __bf16 sA[2][BM * BK];   // 2 x 4 KB
  __shared__ __bf16 sB[2][U_ * BK];   // 2 x 8 KB
  const int m0 = blockIdx.x * BM;
  const int chunk = blockIdx.y;
  const int t = threadIdx.x;
  const int wave = t >> 5, lane = t & 31;
  const int msub = wave & 3, ugrp = wave >> 2;
  const int lrow = lane & 15;
  const int arow = t >> 2, ac0 = (t & 3) * 8;   // A-fill: 8 floats/thread
  const int bu   = t >> 1, be0 = (t & 1) * 16;  // B-fill: 16 bf16/thread
  const int r0 = chunk * RPC;
  v8f acc[4] = {};

  {  // prologue: stage tile 0 into buffer 0
    const float* asrc = A + (size_t)r0 * N_ * N_ + (size_t)(m0 + arow) * N_ + ac0;
    v4f a0 = __builtin_nontemporal_load((const v4f*)asrc);
    v4f a1 = __builtin_nontemporal_load((const v4f*)asrc + 1);
    *(v4bf*)&sA[0][arow * BK + ac0]     = cvt4(a0);
    *(v4bf*)&sA[0][arow * BK + ac0 + 4] = cvt4(a1);
    async_b_copy(yT + (size_t)r0 * U_ * N_, 2u * (unsigned)(bu * N_ + be0),
                 &sB[0][bu * BK + be0]);
    wait_async0();
  }
  __syncthreads();

  for (int kt = 0; kt < TPC; ++kt) {
    const int cur = kt & 1, nxt = cur ^ 1;
    const bool havenext = (kt + 1 < TPC);
    v4f a0 = {}, a1 = {};
    if (havenext) {  // issue next-tile global traffic up front
      const int rn  = r0 + (kt + 1) / NTB;
      const int mmn = ((kt + 1) % NTB) * BK;
      const float* asrc = A + (size_t)rn * N_ * N_ + (size_t)(m0 + arow) * N_ + mmn + ac0;
      a0 = __builtin_nontemporal_load((const v4f*)asrc);
      a1 = __builtin_nontemporal_load((const v4f*)asrc + 1);
      __builtin_prefetch(asrc + 8 * BK, 0, 0);   // stream A ahead (global_prefetch)
      async_b_copy(yT + (size_t)rn * U_ * N_, 2u * (unsigned)(bu * N_ + mmn + be0),
                   &sB[nxt][bu * BK + be0]);
    }
    // compute on current buffers
    v16bf a = afrag_lds(&sA[cur][(msub * 16 + lrow) * BK], lane);
#pragma unroll
    for (int j = 0; j < 4; ++j) {
      v16bf b = bfrag_lds(&sB[cur][(ugrp * 64 + j * 16 + lrow) * BK], lane);
      acc[j] = __builtin_amdgcn_wmma_f32_16x16x32_bf16(false, a, false, b,
                                                       (short)0, acc[j], false, false);
    }
    // publish next buffers
    if (havenext) {
      *(v4bf*)&sA[nxt][arow * BK + ac0]     = cvt4(a0);
      *(v4bf*)&sA[nxt][arow * BK + ac0 + 4] = cvt4(a1);
      wait_async0();
    }
    __syncthreads();
  }

  float* pc = part + (size_t)chunk * N_ * U_;
#pragma unroll
  for (int j = 0; j < 4; ++j) {
    int ucol = ugrp * 64 + j * 16 + lrow;
#pragma unroll
    for (int v = 0; v < 8; ++v) {
      int mrow = m0 + msub * 16 + v + ((lane & 16) ? 8 : 0);
      pc[(size_t)mrow * U_ + ucol] = acc[j][v];
    }
  }
}

// ---- kernel 4: out = relu(sum_c partial[c] + yself + bias) ----
__global__ __launch_bounds__(256) void k_final(const float* __restrict__ part,
                                               const float* __restrict__ yself,
                                               const float* __restrict__ bias,
                                               float* __restrict__ out) {
  int idx = blockIdx.x * 256 + threadIdx.x;
  int u = idx & (U_ - 1);
  float s = yself[idx] + bias[u];
#pragma unroll
  for (int c = 0; c < SPLITK; ++c) s += part[(size_t)c * N_ * U_ + idx];
  out[idx] = s > 0.f ? s : 0.f;
}

extern "C" void kernel_launch(void* const* d_in, const int* in_sizes, int n_in,
                              void* d_out, int out_size, void* d_ws, size_t ws_size,
                              hipStream_t stream) {
  const float* x     = (const float*)d_in[0];  // (1,N,F)
  const float* A     = (const float*)d_in[1];  // (R,N,N)
  const float* bases = (const float*)d_in[2];  // (F,U,B)
  const float* coef  = (const float*)d_in[3];  // (R,B)
  const float* selfk = (const float*)d_in[4];  // (F,U)
  const float* bias  = (const float*)d_in[5];  // (U)
  float* out = (float*)d_out;

  char* ws = (char*)d_ws;
  float*  wrel  = (float*)ws;                                                  // 2 MB
  __bf16* yT    = (__bf16*)(ws + (size_t)R_ * F_ * U_ * 4);                    // 16 MB
  float*  yself = (float*)(ws + (size_t)R_ * F_ * U_ * 4 + (size_t)R_ * U_ * N_ * 2);  // 2 MB
  float*  part  = (float*)((char*)yself + (size_t)N_ * U_ * 4);                // 8 MB

  k_wrel <<<(R_ * F_ * U_) / 256, 256, 0, stream>>>(bases, coef, wrel);
  k_y    <<<dim3(N_ / BM, R_ + 1), 256, 0, stream>>>(x, wrel, selfk, yT, yself);
  k_main <<<dim3(N_ / BM, SPLITK), 256, 0, stream>>>(A, yT, part);
  k_final<<<(N_ * U_) / 256, 256, 0, stream>>>(part, yself, bias, out);
}
